// EncoderBlock_46600395161643
// MI455X (gfx1250) — matrix operand order repaired and checked by reading, not compiled, not run on hardware
//
#include <hip/hip_runtime.h>
#include <hip/hip_bf16.h>

// Problem constants (compile-time, from the reference)
#define BB   8
#define SS   1024
#define DD   1024
#define HH   16
#define DKK  64
#define DFFF 4096
#define MM   (BB * SS)   // 8192 rows

typedef __attribute__((ext_vector_type(16))) __bf16 v16bf;
typedef __attribute__((ext_vector_type(8)))  float  v8f;

#define WMMA_BF16(A_, B_, C_) \
  __builtin_amdgcn_wmma_f32_16x16x32_bf16(false, (A_), false, (B_), (short)0, (C_), false, false)

union Frag16 { v16bf v; uint4 q[2]; };

__device__ __forceinline__ v16bf load_frag(const __bf16* p0, const __bf16* p1) {
  Frag16 f;
  f.q[0] = *reinterpret_cast<const uint4*>(p0);
  f.q[1] = *reinterpret_cast<const uint4*>(p1);
  return f.v;
}
// 16 contiguous bf16 (B-operand pattern: K = e + 16*half, half folded into pointer)
__device__ __forceinline__ v16bf load_frag16(const __bf16* p) { return load_frag(p, p + 8); }

__device__ __forceinline__ void split_bf16(float x, __bf16& hi, __bf16& lo) {
  hi = (__bf16)x;
  lo = (__bf16)(x - (float)hi);
}

// LDS byte address (wave-relative) of a __shared__ object: generic LDS address
// is {SHARED_BASE, offset}; low 32 bits are the LDS address.
__device__ __forceinline__ unsigned lds_addr_of(const void* p) {
  return (unsigned)(unsigned long long)p;
}

// Async global->LDS 16B copy per lane (CDNA5 VGLOBAL async op, ASYNCcnt).
__device__ __forceinline__ void async_copy_b128(unsigned lds_off, const void* gptr) {
  asm volatile("global_load_async_to_lds_b128 %0, %1, off"
               :: "v"(lds_off), "v"(gptr)
               : "memory");
}

// ---------------------------------------------------------------------------
// Weight convert: f32 [K x N] -> transposed bf16 hi/lo [N x K]
// ---------------------------------------------------------------------------
__global__ __launch_bounds__(256) void cvt_wt_kernel(
    const float* __restrict__ w, __bf16* __restrict__ oh, __bf16* __restrict__ ol,
    int K, int N) {
  int idx = blockIdx.x * 256 + threadIdx.x;   // grid sized exactly: K*N % 256 == 0
  int k = idx / N;
  int n = idx - k * N;
  __bf16 hi, lo;
  split_bf16(w[idx], hi, lo);
  size_t o = (size_t)n * (size_t)K + (size_t)k;
  oh[o] = hi;
  ol[o] = lo;
}

// ---------------------------------------------------------------------------
// LayerNorm (ddof=1, scalar gamma/beta, /(std+eps)) -> bf16 hi/lo split
// One 256-thread block per row of D=1024.
// ---------------------------------------------------------------------------
__global__ __launch_bounds__(256) void ln_split_kernel(
    const float* __restrict__ x, const float* __restrict__ g, const float* __restrict__ be,
    __bf16* __restrict__ oh, __bf16* __restrict__ ol) {
  const int row = blockIdx.x;
  const int tid = threadIdx.x;
  const float* xr = x + (size_t)row * DD;
  float v[4];
  float s1 = 0.f, s2 = 0.f;
#pragma unroll
  for (int i = 0; i < 4; ++i) {
    v[i] = xr[tid + i * 256];
    s1 += v[i];
    s2 += v[i] * v[i];
  }
#pragma unroll
  for (int off = 1; off < 32; off <<= 1) {
    s1 += __shfl_xor(s1, off, 32);
    s2 += __shfl_xor(s2, off, 32);
  }
  __shared__ float sh1[8], sh2[8];
  const int wid = tid >> 5;
  if ((tid & 31) == 0) { sh1[wid] = s1; sh2[wid] = s2; }
  __syncthreads();
  s1 = 0.f; s2 = 0.f;
#pragma unroll
  for (int i = 0; i < 8; ++i) { s1 += sh1[i]; s2 += sh2[i]; }
  const float mean = s1 * (1.0f / (float)DD);
  float var = (s2 - (float)DD * mean * mean) * (1.0f / (float)(DD - 1));
  var = fmaxf(var, 0.0f);
  const float a = g[0] / (sqrtf(var) + 1e-6f);
  const float b0 = be[0];
#pragma unroll
  for (int i = 0; i < 4; ++i) {
    float y = a * (v[i] - mean) + b0;
    __bf16 hi, lo;
    split_bf16(y, hi, lo);
    size_t o = (size_t)row * DD + tid + i * 256;
    oh[o] = hi;
    ol[o] = lo;
  }
}

// ---------------------------------------------------------------------------
// Generic bf16x3 split-precision WMMA GEMM with async-LDS B-panel staging.
//   A:  [M x K] bf16 hi/lo, row-major (direct global b128 fragment loads)
//   Bt: [N x K] bf16 hi/lo (weights pre-transposed); the 64xK panel for this
//       block is shared by all 4 waves -> staged into LDS with
//       global_load_async_to_lds_b128, double-buffered on ASYNCcnt.
//   C tile per wave: 64(M) x 64(N); block = 4 waves stacked in M (256x64)
// Epilogues (EPI):
//   0: Oh/Ol = split((acc+bias)*alpha)              row-major [M x N]
//   1: Oh/Ol = split(acc+bias), stored transposed to [B,H,DK,S] (for V)
//   2: Of    = res + acc + bias                     f32 [M x N]
//   3: Oh/Ol = split(relu(acc+bias))                row-major [M x N]
// ---------------------------------------------------------------------------
template <int EPI>
__global__ __launch_bounds__(128) void gemm_bf16x3(
    const __bf16* __restrict__ Ah, const __bf16* __restrict__ Al,
    const __bf16* __restrict__ Bh, const __bf16* __restrict__ Bl,
    const float* __restrict__ bias, const float* __restrict__ res,
    __bf16* __restrict__ Oh, __bf16* __restrict__ Ol, float* __restrict__ Of,
    int Ksz, int Nsz, float alpha) {
  const int tid = threadIdx.x;
  const int lane = tid & 31;
  const int wave = tid >> 5;
  const int l16 = lane & 15;
  const int hf = lane >> 4;
  const int m0 = (blockIdx.x * 4 + wave) * 64;
  const int n0 = blockIdx.y * 64;

  // Double-buffered B panel: [buf][hi/lo][64 rows][32 cols padded to 40]
  __shared__ __align__(16) __bf16 LB[2][2][64][40];

  // Issue one stage: 64 rows x 64B (hi and lo) = 256 x 16B chunks each,
  // 2 chunks per thread per matrix -> 4 async b128 per thread.
  auto issue_stage = [&](int bi, int kk) {
#pragma unroll
    for (int c = 0; c < 2; ++c) {
      const int chunk = tid + c * 128;       // 0..255
      const int row = chunk >> 2;            // 0..63
      const int seg = chunk & 3;             // 16B segment within the 64B row
      const size_t goff = (size_t)(n0 + row) * (size_t)Ksz + kk + seg * 8;
      async_copy_b128(lds_addr_of(&LB[bi][0][row][0]) + seg * 16, Bh + goff);
      async_copy_b128(lds_addr_of(&LB[bi][1][row][0]) + seg * 16, Bl + goff);
    }
  };

  const v8f vz = {0.f, 0.f, 0.f, 0.f, 0.f, 0.f, 0.f, 0.f};
  v8f acc[4][4];
#pragma unroll
  for (int i = 0; i < 4; ++i)
#pragma unroll
    for (int j = 0; j < 4; ++j) acc[i][j] = vz;

  const int nk = Ksz >> 5;
  issue_stage(0, 0);

  for (int i = 0; i < nk; ++i) {
    const int k0 = i << 5;
    if (i + 1 < nk) {
      issue_stage((i + 1) & 1, k0 + 32);
      // previous 4 issued for stage i complete; stage i+1 (4) still in flight
      asm volatile("s_wait_asynccnt 0x4" ::: "memory");
    } else {
      asm volatile("s_wait_asynccnt 0x0" ::: "memory");
    }
    __syncthreads();   // stage-i LDS data visible to all waves

    const int bi = i & 1;
    v16bf bh[4], bl[4];
#pragma unroll
    for (int nt = 0; nt < 4; ++nt) {
      bh[nt] = load_frag(&LB[bi][0][nt * 16 + l16][16 * hf],
                         &LB[bi][0][nt * 16 + l16][16 * hf + 8]);
      bl[nt] = load_frag(&LB[bi][1][nt * 16 + l16][16 * hf],
                         &LB[bi][1][nt * 16 + l16][16 * hf + 8]);
    }
#pragma unroll
    for (int mt = 0; mt < 4; ++mt) {
      const size_t aoff = (size_t)(m0 + mt * 16 + l16) * (size_t)Ksz + k0;
      if (k0 + 32 < Ksz) __builtin_prefetch(Ah + aoff + 32, 0, 1);
      v16bf ah = load_frag(Ah + aoff + 16 * hf, Ah + aoff + 16 * hf + 8);
      v16bf al = load_frag(Al + aoff + 16 * hf, Al + aoff + 16 * hf + 8);
#pragma unroll
      for (int nt = 0; nt < 4; ++nt) {
        acc[mt][nt] = WMMA_BF16(ah, bh[nt], acc[mt][nt]);
        acc[mt][nt] = WMMA_BF16(ah, bl[nt], acc[mt][nt]);
        acc[mt][nt] = WMMA_BF16(al, bh[nt], acc[mt][nt]);
      }
    }
    __syncthreads();   // all waves done with buf bi before it is re-staged
  }

#pragma unroll
  for (int mt = 0; mt < 4; ++mt) {
#pragma unroll
    for (int nt = 0; nt < 4; ++nt) {
      const int n = n0 + nt * 16 + l16;
      const float bv = bias[n];
      if (EPI == 1) {
        union { __bf16 b[8]; uint4 q; } th, tl;
#pragma unroll
        for (int r = 0; r < 8; ++r) {
          float v = (acc[mt][nt][r] + bv) * alpha;
          split_bf16(v, th.b[r], tl.b[r]);
        }
        const int mrow = m0 + mt * 16 + 8 * hf;           // 8 consecutive rows
        const int bbi = mrow >> 10, ssi = mrow & (SS - 1);
        const int hhi = n >> 6, dki = n & (DKK - 1);
        const size_t vb = (((size_t)bbi * HH + hhi) * DKK + dki) * SS + ssi;
        *reinterpret_cast<uint4*>(Oh + vb) = th.q;
        *reinterpret_cast<uint4*>(Ol + vb) = tl.q;
      } else {
#pragma unroll
        for (int r = 0; r < 8; ++r) {
          const int m = m0 + mt * 16 + 8 * hf + r;
          const size_t o = (size_t)m * (size_t)Nsz + n;
          float v = acc[mt][nt][r] + bv;
          if (EPI == 0) {
            v *= alpha;
            __bf16 hi, lo;
            split_bf16(v, hi, lo);
            Oh[o] = hi; Ol[o] = lo;
          }
          if (EPI == 2) { Of[o] = res[o] + v; }
          if (EPI == 3) {
            v = fmaxf(v, 0.0f);
            __bf16 hi, lo;
            split_bf16(v, hi, lo);
            Oh[o] = hi; Ol[o] = lo;
          }
        }
      }
    }
  }
}

// ---------------------------------------------------------------------------
// Flash attention: one wave per (b, h, 16-row q-tile). Q pre-scaled by 1/8.
// Q/K: [M x D] bf16 hi/lo row-major. V: [B,H,DK,S] bf16 hi/lo (transposed).
// Output: attn-out [M x D] bf16 hi/lo.
// ---------------------------------------------------------------------------
__global__ __launch_bounds__(32) void flash_kernel(
    const __bf16* __restrict__ Qh, const __bf16* __restrict__ Ql,
    const __bf16* __restrict__ Kh, const __bf16* __restrict__ Kl,
    const __bf16* __restrict__ Vth, const __bf16* __restrict__ Vtl,
    const int* __restrict__ mask,
    __bf16* __restrict__ AOh, __bf16* __restrict__ AOl) {
  const int lane = threadIdx.x;
  const int l16 = lane & 15;
  const int hf = lane >> 4;
  const int blk = blockIdx.x;               // B*H*(S/16) blocks
  const int qt = blk & 63;
  const int h = (blk >> 6) & (HH - 1);
  const int b = blk >> 10;
  const int m0 = b * SS + qt * 16;

  // Q fragments (A-operand), resident for the whole kernel
  v16bf qfh[2], qfl[2];
  {
    const size_t qr = (size_t)(m0 + l16) * DD + h * DKK;
#pragma unroll
    for (int t = 0; t < 2; ++t) {
      qfh[t] = load_frag(Qh + qr + t * 32 + 16 * hf, Qh + qr + t * 32 + 16 * hf + 8);
      qfl[t] = load_frag(Ql + qr + t * 32 + 16 * hf, Ql + qr + t * 32 + 16 * hf + 8);
    }
  }

  const v8f vz = {0.f, 0.f, 0.f, 0.f, 0.f, 0.f, 0.f, 0.f};
  v8f acc[4] = {vz, vz, vz, vz};
  float mrun[8], lrun[8];
#pragma unroll
  for (int r = 0; r < 8; ++r) { mrun[r] = -1e30f; lrun[r] = 0.0f; }

  __shared__ __align__(16) __bf16 P[16][40];   // 16x32 P tile, padded rows
  const int* mrow = mask + b * SS;

  for (int sk = 0; sk < SS; sk += 32) {
    // --- scores: two 16x16 tiles, K=64 in two 32-chunks, hi/lo x3 ---
    v8f s[2] = {vz, vz};
#pragma unroll
    for (int t = 0; t < 2; ++t) {
      const size_t kr = (size_t)(b * SS + sk + t * 16 + l16) * DD + h * DKK + 16 * hf;
#pragma unroll
      for (int d = 0; d < 2; ++d) {
        v16bf kbh = load_frag16(Kh + kr + d * 32);
        v16bf kbl = load_frag16(Kl + kr + d * 32);
        s[t] = WMMA_BF16(qfh[d], kbh, s[t]);
        s[t] = WMMA_BF16(qfh[d], kbl, s[t]);
        s[t] = WMMA_BF16(qfl[d], kbh, s[t]);
      }
    }
    // --- mask (branchless selects; EXEC stays all-ones) ---
    const int mv0 = mrow[sk + l16];
    const int mv1 = mrow[sk + 16 + l16];
#pragma unroll
    for (int r = 0; r < 8; ++r) {
      s[0][r] = (mv0 == 0) ? -1e9f : s[0][r];
      s[1][r] = (mv1 == 0) ? -1e9f : s[1][r];
    }
    // --- online softmax; row r lives in 16 lanes of one half-wave ---
#pragma unroll
    for (int r = 0; r < 8; ++r) {
      float smax = fmaxf(s[0][r], s[1][r]);
      smax = fmaxf(smax, __shfl_xor(smax, 1, 32));
      smax = fmaxf(smax, __shfl_xor(smax, 2, 32));
      smax = fmaxf(smax, __shfl_xor(smax, 4, 32));
      smax = fmaxf(smax, __shfl_xor(smax, 8, 32));
      const float mnew = fmaxf(mrun[r], smax);
      const float scl = __expf(mrun[r] - mnew);
      const float p0 = __expf(s[0][r] - mnew);
      const float p1 = __expf(s[1][r] - mnew);
      float rs = p0 + p1;
      rs += __shfl_xor(rs, 1, 32);
      rs += __shfl_xor(rs, 2, 32);
      rs += __shfl_xor(rs, 4, 32);
      rs += __shfl_xor(rs, 8, 32);
      lrun[r] = lrun[r] * scl + rs;
      mrun[r] = mnew;
      s[0][r] = p0;
      s[1][r] = p1;
      acc[0][r] *= scl; acc[1][r] *= scl; acc[2][r] *= scl; acc[3][r] *= scl;
    }
    // --- P: C-layout -> LDS -> A-fragment layout ---
#pragma unroll
    for (int r = 0; r < 8; ++r) {
      P[r + 8 * hf][l16] = (__bf16)s[0][r];
      P[r + 8 * hf][16 + l16] = (__bf16)s[1][r];
    }
    __syncthreads();
    v16bf pf = load_frag(&P[l16][16 * hf], &P[l16][16 * hf + 8]);
    __syncthreads();
    // --- P @ V (V hi/lo) ---
#pragma unroll
    for (int nt = 0; nt < 4; ++nt) {
      const size_t vr = (((size_t)b * HH + h) * DKK + nt * 16 + l16) * SS + sk + 16 * hf;
      v16bf vh = load_frag16(Vth + vr);
      v16bf vl = load_frag16(Vtl + vr);
      acc[nt] = WMMA_BF16(pf, vh, acc[nt]);
      acc[nt] = WMMA_BF16(pf, vl, acc[nt]);
    }
  }
  // --- finalize: divide by row sums, split-store to [M x D] ---
#pragma unroll
  for (int r = 0; r < 8; ++r) {
    const float inv = 1.0f / lrun[r];
    const int m = m0 + 8 * hf + r;
#pragma unroll
    for (int nt = 0; nt < 4; ++nt) {
      float v = acc[nt][r] * inv;
      __bf16 hi, lo;
      split_bf16(v, hi, lo);
      const size_t o = (size_t)m * DD + h * DKK + nt * 16 + l16;
      AOh[o] = hi;
      AOl[o] = lo;
    }
  }
}

// ---------------------------------------------------------------------------
// Launch: needs ~240 MiB of d_ws.
// ---------------------------------------------------------------------------
extern "C" void kernel_launch(void* const* d_in, const int* in_sizes, int n_in,
                              void* d_out, int out_size, void* d_ws, size_t ws_size,
                              hipStream_t stream) {
  (void)in_sizes; (void)n_in; (void)out_size; (void)ws_size;
  const float* x   = (const float*)d_in[0];
  const int*  mask = (const int*)d_in[1];
  const float* wq = (const float*)d_in[2];
  const float* bq = (const float*)d_in[3];
  const float* wk = (const float*)d_in[4];
  const float* bk = (const float*)d_in[5];
  const float* wv = (const float*)d_in[6];
  const float* bv = (const float*)d_in[7];
  const float* wo = (const float*)d_in[8];
  const float* bo = (const float*)d_in[9];
  const float* w1 = (const float*)d_in[10];
  const float* b1 = (const float*)d_in[11];
  const float* w2 = (const float*)d_in[12];
  const float* b2 = (const float*)d_in[13];
  const float* g1  = (const float*)d_in[14];
  const float* be1 = (const float*)d_in[15];
  const float* g2  = (const float*)d_in[16];
  const float* be2 = (const float*)d_in[17];

  char* ws = (char*)d_ws;
  const size_t MB = 1ull << 20;
  auto BF = [&](size_t off) { return (__bf16*)(ws + off); };

  // Workspace layout (MiB offsets):
  //   0..48   : transposed weight splits
  //   48..80  : LN output hi/lo (reused for LN2)
  //   80..208 : Q/K/Vt/AO hi/lo  (reused as FFN hidden hi/lo)
  //   208..240: x1 (f32 residual stream)
  __bf16 *wqt_h = BF(0 * MB),  *wqt_l = BF(2 * MB);
  __bf16 *wkt_h = BF(4 * MB),  *wkt_l = BF(6 * MB);
  __bf16 *wvt_h = BF(8 * MB),  *wvt_l = BF(10 * MB);
  __bf16 *wot_h = BF(12 * MB), *wot_l = BF(14 * MB);
  __bf16 *w1t_h = BF(16 * MB), *w1t_l = BF(24 * MB);
  __bf16 *w2t_h = BF(32 * MB), *w2t_l = BF(40 * MB);
  __bf16 *Xh = BF(48 * MB), *Xl = BF(64 * MB);
  __bf16 *Qh = BF(80 * MB),  *Ql = BF(96 * MB);
  __bf16 *Kh = BF(112 * MB), *Kl = BF(128 * MB);
  __bf16 *Vth = BF(144 * MB), *Vtl = BF(160 * MB);
  __bf16 *AOh = BF(176 * MB), *AOl = BF(192 * MB);
  float  *X1 = (float*)(ws + 208 * MB);
  __bf16 *A1h = BF(80 * MB), *A1l = BF(144 * MB);   // aliases Q/K and Vt/AO (dead by then)

  // 1) weight conversion (transposed hi/lo splits)
  cvt_wt_kernel<<<4096, 256, 0, stream>>>(wq, wqt_h, wqt_l, DD, DD);
  cvt_wt_kernel<<<4096, 256, 0, stream>>>(wk, wkt_h, wkt_l, DD, DD);
  cvt_wt_kernel<<<4096, 256, 0, stream>>>(wv, wvt_h, wvt_l, DD, DD);
  cvt_wt_kernel<<<4096, 256, 0, stream>>>(wo, wot_h, wot_l, DD, DD);
  cvt_wt_kernel<<<16384, 256, 0, stream>>>(w1, w1t_h, w1t_l, DD, DFFF);
  cvt_wt_kernel<<<16384, 256, 0, stream>>>(w2, w2t_h, w2t_l, DFFF, DD);

  // 2) LN1
  ln_split_kernel<<<MM, 256, 0, stream>>>(x, g1, be1, Xh, Xl);

  // 3) Q (pre-scaled by 1/sqrt(DK)), K, V (transposed store)
  dim3 gD(MM / 256, DD / 64);
  gemm_bf16x3<0><<<gD, 128, 0, stream>>>(Xh, Xl, wqt_h, wqt_l, bq, nullptr,
                                         Qh, Ql, nullptr, DD, DD, 0.125f);
  gemm_bf16x3<0><<<gD, 128, 0, stream>>>(Xh, Xl, wkt_h, wkt_l, bk, nullptr,
                                         Kh, Kl, nullptr, DD, DD, 1.0f);
  gemm_bf16x3<1><<<gD, 128, 0, stream>>>(Xh, Xl, wvt_h, wvt_l, bv, nullptr,
                                         Vth, Vtl, nullptr, DD, DD, 1.0f);

  // 4) flash attention
  flash_kernel<<<BB * HH * (SS / 16), 32, 0, stream>>>(Qh, Ql, Kh, Kl, Vth, Vtl,
                                                       mask, AOh, AOl);

  // 5) output projection + residual -> x1
  gemm_bf16x3<2><<<gD, 128, 0, stream>>>(AOh, AOl, wot_h, wot_l, bo, x,
                                         nullptr, nullptr, X1, DD, DD, 1.0f);

  // 6) LN2
  ln_split_kernel<<<MM, 256, 0, stream>>>(X1, g2, be2, Xh, Xl);

  // 7) FFN1 (ReLU, split)
  dim3 gF(MM / 256, DFFF / 64);
  gemm_bf16x3<3><<<gF, 128, 0, stream>>>(Xh, Xl, w1t_h, w1t_l, b1, nullptr,
                                         A1h, A1l, nullptr, DD, DFFF, 1.0f);

  // 8) FFN2 + residual -> d_out (f32)
  gemm_bf16x3<2><<<gD, 128, 0, stream>>>(A1h, A1l, w2t_h, w2t_l, b2, X1,
                                         nullptr, nullptr, (float*)d_out, DFFF, DD, 1.0f);
}
